// NMTModel_62852551410155
// MI455X (gfx1250) — matrix operand order repaired and checked by reading, not compiled, not run on hardware
//
#include <hip/hip_runtime.h>

typedef __attribute__((ext_vector_type(16))) __bf16 v16bf;
typedef __attribute__((ext_vector_type(8)))  float  v8f;

constexpr int Bz = 32, Sz = 64, Tz = 48, Ez = 512, Hz = 1024, Vz = 32000;
constexpr int G4 = 4 * Hz;                  // 4096
constexpr int ROWS = (Tz - 1) * Bz;         // 1504 rows scored
constexpr int CH = 512;                     // vocab chunk width
constexpr int NCHUNK = (Vz + CH - 1) / CH;  // 63

union ABfrag { uint4 u[2]; v16bf v; };

__device__ __forceinline__ float sigf(float x) { return 1.0f / (1.0f + expf(-x)); }

// ---------------------------------------------------------------------------
// Generic bf16 WMMA GEMM:  Cout[M,N] = A[M,K] @ Bw[N,K]^T (+ Cinit + bias)
// Each wave computes a 16x64 output strip (4 accumulators), K in steps of 32.
// ---------------------------------------------------------------------------
__global__ __launch_bounds__(256) void gemm_bf16_kernel(
    const __bf16* __restrict__ A, int lda,
    const __bf16* __restrict__ Bw, int ldb,
    const float* __restrict__ Cinit, int ldci,
    const float* __restrict__ bias,
    float* __restrict__ Cout, int ldc,
    __bf16* __restrict__ Cout16, int ldc16,
    int M, int N, int K)
{
  const int wave = blockIdx.x * (blockDim.x >> 5) + (threadIdx.x >> 5);
  const int nGroups = N >> 6;
  if (wave >= (M >> 4) * nGroups) return;
  const int mBase = (wave / nGroups) << 4;
  const int nBase = (wave % nGroups) << 6;
  const int lane = threadIdx.x & 31;
  const int hl   = lane >> 4;      // 0 or 1 (half-wave)
  const int l16  = lane & 15;
  const int aRow = mBase + l16;
  const int kOffA = hl << 3;       // A: K offset 0 / 8
  const int kGrpB = hl << 3;       // B: K offset 0 / 8
  const int rowOff = hl << 3;      // C: row offset 0 / 8

  v8f acc[4];
#pragma unroll
  for (int j = 0; j < 4; ++j) {
    v8f c = {};
    const int col = nBase + (j << 4) + l16;
    const float bv = bias ? bias[col] : 0.0f;
#pragma unroll
    for (int r = 0; r < 8; ++r) {
      float v = bv;
      if (Cinit) v += Cinit[(size_t)(mBase + rowOff + r) * ldci + col];
      c[r] = v;
    }
    acc[j] = c;
  }

  for (int kb = 0; kb < K; kb += 32) {
    ABfrag a;
    const __bf16* ap = A + (size_t)aRow * lda + kb + kOffA;
    a.u[0] = *(const uint4*)(ap);        // K = kb+kOff .. +7
    a.u[1] = *(const uint4*)(ap + 16);   // K = kb+16+kOff .. +7
#pragma unroll
    for (int j = 0; j < 4; ++j) {
      const int nCol = nBase + (j << 4) + l16;
      ABfrag b;
      const __bf16* bp = Bw + (size_t)nCol * ldb + kb + kGrpB;
      b.u[0] = *(const uint4*)(bp);
      b.u[1] = *(const uint4*)(bp + 16);
      acc[j] = __builtin_amdgcn_wmma_f32_16x16x32_bf16(
          false, a.v, false, b.v, (short)0, acc[j], false, false);
    }
  }

#pragma unroll
  for (int j = 0; j < 4; ++j) {
    const int col = nBase + (j << 4) + l16;
#pragma unroll
    for (int r = 0; r < 8; ++r) {
      const int row = mBase + rowOff + r;
      const float v = acc[j][r];
      Cout[(size_t)row * ldc + col] = v;
      if (Cout16) Cout16[(size_t)row * ldc16 + col] = (__bf16)v;
    }
  }
}

// ---------------------------------------------------------------------------
// LSTM cell elementwise (gates pre-summed, order i,f,g,o).
// ---------------------------------------------------------------------------
__global__ __launch_bounds__(256) void lstm_cell_kernel(
    const float* __restrict__ gates,   // (B,4H)
    float* __restrict__ c32,           // (B*H) in/out
    float* __restrict__ h32,           // (B*H) out
    __bf16* __restrict__ h16a, int lda16,
    __bf16* __restrict__ h16b, int ldb16)
{
  const int idx = blockIdx.x * blockDim.x + threadIdx.x;
  if (idx >= Bz * Hz) return;
  const int b = idx >> 10;         // /Hz
  const int k = idx & (Hz - 1);
  const float* g = gates + (size_t)b * G4;
  const float gi = sigf(g[k]);
  const float gf = sigf(g[Hz + k]);
  const float gg = tanhf(g[2 * Hz + k]);
  const float go = sigf(g[3 * Hz + k]);
  const float c = gf * c32[idx] + gi * gg;
  c32[idx] = c;
  const float hh = go * tanhf(c);
  h32[idx] = hh;
  if (h16a) h16a[(size_t)b * lda16 + k] = (__bf16)hh;
  if (h16b) h16b[(size_t)b * ldb16 + k] = (__bf16)hh;
}

// ---------------------------------------------------------------------------
// Attention: scores = h . enc_att[b,s,:], softmax over S, ctx = attn @ enc_h.
// One block per batch element. Writes ctx bf16 into combIn[:, H:3H].
// ---------------------------------------------------------------------------
__global__ __launch_bounds__(256) void attention_kernel(
    const float* __restrict__ h,       // (B,H)
    const float* __restrict__ encAtt,  // (B,S,H)
    const __bf16* __restrict__ encH,   // (B,S,2H)
    __bf16* __restrict__ ctxOut)       // combIn + H, row stride 3H
{
  __shared__ float part[Sz][4];
  __shared__ float attn[Sz];
  __shared__ float mx, sm;
  const int b = blockIdx.x;
  const int tid = threadIdx.x;
  const int s = tid >> 2, q = tid & 3;
  const float* hp = h + (size_t)b * Hz + q * 256;
  const float* ep = encAtt + ((size_t)(b * Sz + s)) * Hz + q * 256;
  float acc = 0.0f;
  for (int k = 0; k < 256; ++k) acc += hp[k] * ep[k];
  part[s][q] = acc;
  __syncthreads();
  if (tid < Sz) attn[tid] = part[tid][0] + part[tid][1] + part[tid][2] + part[tid][3];
  __syncthreads();
  if (tid == 0) {
    float m = -INFINITY;
    for (int i = 0; i < Sz; ++i) m = fmaxf(m, attn[i]);
    float ss = 0.0f;
    for (int i = 0; i < Sz; ++i) ss += expf(attn[i] - m);
    mx = m; sm = ss;
  }
  __syncthreads();
  if (tid < Sz) attn[tid] = expf(attn[tid] - mx) / sm;
  __syncthreads();
  for (int d = tid; d < 2 * Hz; d += 256) {
    float a = 0.0f;
    const __bf16* e2 = encH + (size_t)b * Sz * 2 * Hz + d;
    for (int s2 = 0; s2 < Sz; ++s2) a += attn[s2] * (float)e2[(size_t)s2 * 2 * Hz];
    ctxOut[(size_t)b * 3 * Hz + d] = (__bf16)a;
  }
}

// ---------------------------------------------------------------------------
// Projection pass 1: WMMA 16x512 logit tile -> LDS, per-row chunk max /
// sum-exp statistics + gold logit capture.  grid = (ROWS/16, NCHUNK).
// ---------------------------------------------------------------------------
__global__ __launch_bounds__(256) void proj_pass1_kernel(
    const __bf16* __restrict__ outs16,  // (T*B, H)
    const __bf16* __restrict__ wproj,   // (V, H)
    const int* __restrict__ tgt,        // (B, T)
    float* __restrict__ cmax, float* __restrict__ csum,
    float* __restrict__ gold)
{
  __shared__ float tile[16][CH];
  __shared__ float red[16][17];
  __shared__ float rmax[16];

  const int mBase = blockIdx.x << 4;
  const int chunk = blockIdx.y;
  const int cstart = chunk * CH;
  const int waveId = threadIdx.x >> 5;
  const int lane = threadIdx.x & 31;
  const int hl = lane >> 4;
  const int l16 = lane & 15;
  const int nBase = waveId << 6;
  const int aRow = mBase + l16;
  const int kOffA = hl << 3;
  const int kGrpB = hl << 3;
  const int rowOff = hl << 3;

  v8f acc[4] = {};
  for (int kb = 0; kb < Hz; kb += 32) {
    ABfrag a;
    const __bf16* ap = outs16 + (size_t)aRow * Hz + kb + kOffA;
    a.u[0] = *(const uint4*)(ap);
    a.u[1] = *(const uint4*)(ap + 16);
#pragma unroll
    for (int j = 0; j < 4; ++j) {
      const int vc = cstart + nBase + (j << 4) + l16;
      ABfrag b;
      if (vc < Vz) {
        const __bf16* bp = wproj + (size_t)vc * Hz + kb + kGrpB;
        b.u[0] = *(const uint4*)(bp);
        b.u[1] = *(const uint4*)(bp + 16);
      } else {
        b.u[0] = make_uint4(0u, 0u, 0u, 0u);
        b.u[1] = make_uint4(0u, 0u, 0u, 0u);
      }
      acc[j] = __builtin_amdgcn_wmma_f32_16x16x32_bf16(
          false, a.v, false, b.v, (short)0, acc[j], false, false);
    }
  }
#pragma unroll
  for (int j = 0; j < 4; ++j)
#pragma unroll
    for (int r = 0; r < 8; ++r)
      tile[rowOff + r][nBase + (j << 4) + l16] = acc[j][r];
  __syncthreads();

  const int r  = threadIdx.x >> 4;
  const int c0 = threadIdx.x & 15;
  float m = -INFINITY;
  for (int c = c0; c < CH; c += 16)
    if (cstart + c < Vz) m = fmaxf(m, tile[r][c]);
  red[r][c0] = m;
  __syncthreads();
  if (c0 == 0) {
    float mm = -INFINITY;
    for (int i = 0; i < 16; ++i) mm = fmaxf(mm, red[r][i]);
    rmax[r] = mm;
  }
  __syncthreads();
  const float mm = rmax[r];
  float s = 0.0f;
  for (int c = c0; c < CH; c += 16)
    if (cstart + c < Vz) s += expf(tile[r][c] - mm);
  red[r][c0] = s;
  __syncthreads();
  if (c0 == 0) {
    float ss = 0.0f;
    for (int i = 0; i < 16; ++i) ss += red[r][i];
    const int gr = mBase + r;
    cmax[(size_t)gr * NCHUNK + chunk] = mm;
    csum[(size_t)gr * NCHUNK + chunk] = ss;
    const int tt = gr >> 5;   // gr / B
    const int bb = gr & 31;   // gr % B
    const int g = tgt[bb * Tz + tt + 1];
    if (g >= cstart && g < cstart + CH) gold[gr] = tile[r][g - cstart];
  }
}

// Projection pass 2: combine chunk stats via log-sum-exp, atomicAdd per batch.
__global__ void proj_pass2_kernel(const float* __restrict__ cmax,
                                  const float* __restrict__ csum,
                                  const float* __restrict__ gold,
                                  float* __restrict__ out)
{
  const int gr = blockIdx.x * blockDim.x + threadIdx.x;
  if (gr >= ROWS) return;
  float gm = -INFINITY;
  for (int j = 0; j < NCHUNK; ++j) gm = fmaxf(gm, cmax[(size_t)gr * NCHUNK + j]);
  float s = 0.0f;
  for (int j = 0; j < NCHUNK; ++j)
    s += csum[(size_t)gr * NCHUNK + j] * expf(cmax[(size_t)gr * NCHUNK + j] - gm);
  const float val = gold[gr] - gm - logf(s);
  atomicAdd(&out[gr & 31], val);
}

// ---------------------------------------------------------------------------
// Utility kernels.
// ---------------------------------------------------------------------------
__global__ void gather_emb_kernel(const float* __restrict__ emb,  // (V,E)
                                  const int* __restrict__ seq,    // (B,L)
                                  __bf16* __restrict__ out,       // (L,B,E)
                                  int L)
{
  const long idx = (long)blockIdx.x * blockDim.x + threadIdx.x;
  if (idx >= (long)L * Bz * Ez) return;
  const int e = (int)(idx & (Ez - 1));
  const long r = idx >> 9;            // /Ez
  const int b = (int)(r % Bz);
  const int l = (int)(r / Bz);
  const int tok = seq[b * L + l];
  out[idx] = (__bf16)emb[(size_t)tok * Ez + e];
}

__global__ void convert2d_kernel(const float* __restrict__ src, int lds,
                                 __bf16* __restrict__ dst, int ldd,
                                 long rows, int cols)
{
  const long idx = (long)blockIdx.x * blockDim.x + threadIdx.x;
  if (idx >= rows * (long)cols) return;
  const int c = (int)(idx % cols);
  const long r = idx / cols;
  dst[r * (long)ldd + c] = (__bf16)src[r * (long)lds + c];
}

__global__ void bias_sum_kernel(const float* a, const float* b, float* o, int n)
{
  const int i = blockIdx.x * blockDim.x + threadIdx.x;
  if (i < n) o[i] = a[i] + b[i];
}

__global__ void zero_f32_kernel(float* p, long n)
{
  const long i = (long)blockIdx.x * blockDim.x + threadIdx.x;
  if (i < n) p[i] = 0.0f;
}

__global__ void zero_bf16_kernel(__bf16* p, long n)
{
  const long i = (long)blockIdx.x * blockDim.x + threadIdx.x;
  if (i < n) p[i] = (__bf16)0.0f;
}

// ---------------------------------------------------------------------------
extern "C" void kernel_launch(void* const* d_in, const int* in_sizes, int n_in,
                              void* d_out, int out_size, void* d_ws, size_t ws_size,
                              hipStream_t stream)
{
  (void)in_sizes; (void)n_in; (void)out_size; (void)ws_size;
  const int*   seqSrc = (const int*)d_in[0];
  const int*   seqTgt = (const int*)d_in[1];
  const float* srcEmb = (const float*)d_in[2];
  const float* tgtEmb = (const float*)d_in[3];
  const float* eWihF  = (const float*)d_in[4];
  const float* eWhhF  = (const float*)d_in[5];
  const float* eBihF  = (const float*)d_in[6];
  const float* eBhhF  = (const float*)d_in[7];
  const float* eWihB  = (const float*)d_in[8];
  const float* eWhhB  = (const float*)d_in[9];
  const float* eBihB  = (const float*)d_in[10];
  const float* eBhhB  = (const float*)d_in[11];
  const float* dWih   = (const float*)d_in[12];
  const float* dWhh   = (const float*)d_in[13];
  const float* dBih   = (const float*)d_in[14];
  const float* dBhh   = (const float*)d_in[15];
  const float* WinitH = (const float*)d_in[16];
  const float* WinitC = (const float*)d_in[17];
  const float* Watt   = (const float*)d_in[18];
  const float* Wcomb  = (const float*)d_in[19];
  const float* Wproj  = (const float*)d_in[20];

  size_t off = 0;
  char* base = (char*)d_ws;
  auto a16 = [&](size_t n) { __bf16* p = (__bf16*)(base + off);
                             off = (off + 2 * n + 255) & ~(size_t)255; return p; };
  auto a32 = [&](size_t n) { float* p = (float*)(base + off);
                             off = (off + 4 * n + 255) & ~(size_t)255; return p; };

  // bf16 buffers
  __bf16* wEncWihF = a16((size_t)G4 * Ez);
  __bf16* wEncWhhF = a16((size_t)G4 * Hz);
  __bf16* wEncWihB = a16((size_t)G4 * Ez);
  __bf16* wEncWhhB = a16((size_t)G4 * Hz);
  __bf16* wDecWih  = a16((size_t)G4 * (Ez + Hz));
  __bf16* wDecRec  = a16((size_t)G4 * 2 * Hz);   // [Wih[:,E:] | Whh]
  __bf16* wInitH16 = a16((size_t)Hz * 2 * Hz);
  __bf16* wInitC16 = a16((size_t)Hz * 2 * Hz);
  __bf16* wAtt16   = a16((size_t)Hz * 2 * Hz);
  __bf16* wComb16  = a16((size_t)Hz * 3 * Hz);
  __bf16* wProj16  = a16((size_t)Vz * Hz);
  __bf16* xSrc16   = a16((size_t)Sz * Bz * Ez);
  __bf16* xTgt16   = a16((size_t)Tz * Bz * Ez);
  __bf16* encH16   = a16((size_t)Bz * Sz * 2 * Hz);
  __bf16* lastH16  = a16((size_t)Bz * 2 * Hz);   // [hf|hb], also recurrent A
  __bf16* lastC16  = a16((size_t)Bz * 2 * Hz);
  __bf16* aRec     = a16((size_t)Bz * 2 * Hz);   // [prev_out | h]
  __bf16* combIn   = a16((size_t)Bz * 3 * Hz);   // [h | ctx]
  __bf16* outs16   = a16((size_t)Tz * Bz * Hz);

  // f32 buffers
  float* gatesXF   = a32((size_t)Sz * Bz * G4);
  float* gatesXB   = a32((size_t)Sz * Bz * G4);
  float* gatesXD   = a32((size_t)Tz * Bz * G4);
  float* gatesStep = a32((size_t)Bz * G4);
  float* states    = a32((size_t)6 * Bz * Hz);
  float* hF = states;
  float* cF = states + 1 * Bz * Hz;
  float* hB = states + 2 * Bz * Hz;
  float* cB = states + 3 * Bz * Hz;
  float* dH = states + 4 * Bz * Hz;
  float* dC = states + 5 * Bz * Hz;
  float* encAtt = a32((size_t)Bz * Sz * Hz);
  float* outs32 = a32((size_t)Tz * Bz * Hz);
  float* bsumF = a32(G4);
  float* bsumB = a32(G4);
  float* bsumD = a32(G4);
  float* cmax  = a32((size_t)ROWS * NCHUNK);
  float* csum  = a32((size_t)ROWS * NCHUNK);
  float* goldv = a32((size_t)ROWS);

  auto gemm = [&](const __bf16* A, int lda, const __bf16* Bw, int ldb,
                  const float* Ci, int ldci, const float* bias,
                  float* C, int ldc, __bf16* C16, int ldc16,
                  int M, int N, int K) {
    const int waves = (M >> 4) * (N >> 6);
    const int blocks = (waves + 7) >> 3;
    gemm_bf16_kernel<<<blocks, 256, 0, stream>>>(A, lda, Bw, ldb, Ci, ldci, bias,
                                                 C, ldc, C16, ldc16, M, N, K);
  };
  auto conv = [&](const float* s, int lds, __bf16* d, int ldd, long rows, int cols) {
    const long n = rows * (long)cols;
    convert2d_kernel<<<(int)((n + 255) >> 8), 256, 0, stream>>>(s, lds, d, ldd, rows, cols);
  };

  // ---- init / zero state ----
  zero_f32_kernel<<<(6 * Bz * Hz + 255) / 256, 256, 0, stream>>>(states, 6L * Bz * Hz);
  zero_bf16_kernel<<<(Bz * 2 * Hz + 255) / 256, 256, 0, stream>>>(lastH16, (long)Bz * 2 * Hz);
  zero_bf16_kernel<<<(Bz * 2 * Hz + 255) / 256, 256, 0, stream>>>(aRec, (long)Bz * 2 * Hz);
  zero_f32_kernel<<<1, 256, 0, stream>>>((float*)d_out, Bz);

  // ---- weights -> bf16 ----
  conv(eWihF, Ez, wEncWihF, Ez, G4, Ez);
  conv(eWhhF, Hz, wEncWhhF, Hz, G4, Hz);
  conv(eWihB, Ez, wEncWihB, Ez, G4, Ez);
  conv(eWhhB, Hz, wEncWhhB, Hz, G4, Hz);
  conv(dWih, Ez + Hz, wDecWih, Ez + Hz, G4, Ez + Hz);
  conv(dWih + Ez, Ez + Hz, wDecRec, 2 * Hz, G4, Hz);       // prev_out part
  conv(dWhh, Hz, wDecRec + Hz, 2 * Hz, G4, Hz);            // h part
  conv(WinitH, 2 * Hz, wInitH16, 2 * Hz, Hz, 2 * Hz);
  conv(WinitC, 2 * Hz, wInitC16, 2 * Hz, Hz, 2 * Hz);
  conv(Watt, 2 * Hz, wAtt16, 2 * Hz, Hz, 2 * Hz);
  conv(Wcomb, 3 * Hz, wComb16, 3 * Hz, Hz, 3 * Hz);
  conv(Wproj, Hz, wProj16, Hz, Vz, Hz);

  // ---- fold biases ----
  bias_sum_kernel<<<(G4 + 255) / 256, 256, 0, stream>>>(eBihF, eBhhF, bsumF, G4);
  bias_sum_kernel<<<(G4 + 255) / 256, 256, 0, stream>>>(eBihB, eBhhB, bsumB, G4);
  bias_sum_kernel<<<(G4 + 255) / 256, 256, 0, stream>>>(dBih, dBhh, bsumD, G4);

  // ---- embedding gathers (time-major, bf16) ----
  {
    long n = (long)Sz * Bz * Ez;
    gather_emb_kernel<<<(int)((n + 255) >> 8), 256, 0, stream>>>(srcEmb, seqSrc, xSrc16, Sz);
    n = (long)Tz * Bz * Ez;
    gather_emb_kernel<<<(int)((n + 255) >> 8), 256, 0, stream>>>(tgtEmb, seqTgt, xTgt16, Tz);
  }

  // ---- hoisted input-gate GEMMs: x @ W_ih^T + (b_ih + b_hh) ----
  gemm(xSrc16, Ez, wEncWihF, Ez, nullptr, 0, bsumF, gatesXF, G4, nullptr, 0, Sz * Bz, G4, Ez);
  gemm(xSrc16, Ez, wEncWihB, Ez, nullptr, 0, bsumB, gatesXB, G4, nullptr, 0, Sz * Bz, G4, Ez);
  gemm(xTgt16, Ez, wDecWih, Ez + Hz, nullptr, 0, bsumD, gatesXD, G4, nullptr, 0, Tz * Bz, G4, Ez);

  // ---- bidirectional encoder scan ----
  for (int t = 0; t < Sz; ++t) {
    // forward: gates = gatesXF[t] + h_f @ Whh_f^T
    gemm(lastH16, 2 * Hz, wEncWhhF, Hz, gatesXF + (size_t)t * Bz * G4, G4, nullptr,
         gatesStep, G4, nullptr, 0, Bz, G4, Hz);
    lstm_cell_kernel<<<(Bz * Hz) / 256, 256, 0, stream>>>(
        gatesStep, cF, hF, lastH16, 2 * Hz, encH16 + (size_t)t * 2 * Hz, Sz * 2 * Hz);
    // backward: source position s = S-1-t
    const int s = Sz - 1 - t;
    gemm(lastH16 + Hz, 2 * Hz, wEncWhhB, Hz, gatesXB + (size_t)s * Bz * G4, G4, nullptr,
         gatesStep, G4, nullptr, 0, Bz, G4, Hz);
    lstm_cell_kernel<<<(Bz * Hz) / 256, 256, 0, stream>>>(
        gatesStep, cB, hB, lastH16 + Hz, 2 * Hz, encH16 + (size_t)s * 2 * Hz + Hz, Sz * 2 * Hz);
  }

  // last cell -> bf16 [cf|cb]
  conv(cF, Hz, lastC16, 2 * Hz, Bz, Hz);
  conv(cB, Hz, lastC16 + Hz, 2 * Hz, Bz, Hz);

  // decoder init states + attention key precompute
  gemm(lastH16, 2 * Hz, wInitH16, 2 * Hz, nullptr, 0, nullptr,
       dH, Hz, aRec + Hz, 2 * Hz, Bz, Hz, 2 * Hz);
  gemm(lastC16, 2 * Hz, wInitC16, 2 * Hz, nullptr, 0, nullptr,
       dC, Hz, nullptr, 0, Bz, Hz, 2 * Hz);
  gemm(encH16, 2 * Hz, wAtt16, 2 * Hz, nullptr, 0, nullptr,
       encAtt, Hz, nullptr, 0, Bz * Sz, Hz, 2 * Hz);

  // ---- decoder scan ----
  for (int t = 0; t < Tz; ++t) {
    // gates = gatesXD[t] + [prev_out|h] @ [Wih[:,E:]|Whh]^T
    gemm(aRec, 2 * Hz, wDecRec, 2 * Hz, gatesXD + (size_t)t * Bz * G4, G4, nullptr,
         gatesStep, G4, nullptr, 0, Bz, G4, 2 * Hz);
    lstm_cell_kernel<<<(Bz * Hz) / 256, 256, 0, stream>>>(
        gatesStep, dC, dH, combIn, 3 * Hz, aRec + Hz, 2 * Hz);
    attention_kernel<<<Bz, 256, 0, stream>>>(dH, encAtt, encH16, combIn + Hz);
    // out = [h|ctx] @ Wcomb^T ; bf16 copy becomes next prev_out
    gemm(combIn, 3 * Hz, wComb16, 3 * Hz, nullptr, 0, nullptr,
         outs32 + (size_t)t * Bz * Hz, Hz, aRec, 2 * Hz, Bz, Hz, 3 * Hz);
  }

  // ---- fused vocab projection + log-softmax + gold gather ----
  conv(outs32, Hz, outs16, Hz, (long)Tz * Bz, Hz);
  proj_pass1_kernel<<<dim3(ROWS / 16, NCHUNK), 256, 0, stream>>>(
      outs16, wProj16, seqTgt, cmax, csum, goldv);
  proj_pass2_kernel<<<(ROWS + 255) / 256, 256, 0, stream>>>(
      cmax, csum, goldv, (float*)d_out);
}
